// GCNModelVAE_36361193128417
// MI455X (gfx1250) — matrix operand order repaired and checked by reading, not compiled, not run on hardware
//
#include <hip/hip_runtime.h>
#include <hip/hip_bf16.h>
#include <cstdint>

// ---------------------------------------------------------------------------
// GCN-VAE forward for MI455X (gfx1250), bf16 WMMA (v_wmma_f32_16x16x32_bf16),
// wave32, 8-wave workgroups, double-buffered LDS tiles, async global->LDS
// staging (ASYNCcnt) where the operand is already bf16.
//
// Restructured math (exact in real arithmetic):
//   T1   = x @ W1                      (write bf16 transposed [256][8192])
//   h1   = relu(adj @ T1)              (write bf16 transposed [256][8192])
//   Amat = adj @ h1                    (write bf16 row-major  [8192][256])
//   P    = Amat @ [W2|W3|W4|W5]        (fp32 [8192][512] = mu|lv|cmu|clv)
//   group evidence over single segment -> gmu[128], glogvar[128]
//   z    = eps_z * exp(lv) + mu
//   U    = z @ Wl[0:128,:] + c0        (c0 folds class_lat @ Wl[128:,:] + bl)
//   recon= U @ U^T
// ---------------------------------------------------------------------------

#if defined(__HIP_DEVICE_COMPILE__) &&                                         \
    __has_builtin(__builtin_amdgcn_global_load_async_to_lds_b128) &&           \
    __has_builtin(__builtin_amdgcn_s_wait_asynccnt)
#define HAVE_ASYNC 1
#else
#define HAVE_ASYNC 0
#endif

typedef __attribute__((ext_vector_type(16))) __bf16 v16bf;
typedef __attribute__((ext_vector_type(2)))  __bf16 v2bf;
typedef __attribute__((ext_vector_type(8)))  float  v8f;
typedef __attribute__((ext_vector_type(2)))  float  v2f;

union FragU { unsigned int u[8]; v16bf v; };

// Pack two f32 -> one dword of bf16x2 via a vector convert; the backend
// pattern-matches this to a single v_cvt_pk_bf16_f32 (no scalar reassembly).
__device__ __forceinline__ unsigned int pk2bf(float lo, float hi) {
  v2f t; t.x = lo; t.y = hi;
  return __builtin_bit_cast(unsigned int, __builtin_convertvector(t, v2bf));
}
// Scalar bf16 store without a ushort zext round-trip.
__device__ __forceinline__ void store_bf16(unsigned short* p, float v) {
  *(__bf16*)p = (__bf16)v;
}

#if HAVE_ASYNC
// The builtin's parameters are pointers to 16-byte int vectors in the global
// (as1) and LDS (as3) address spaces. Casts go through uintptr_t: as(1) is
// numerically identical to flat for global memory; as(3) is the low 32 bits
// of the flat LDS aperture address.
typedef __attribute__((__vector_size__(4 * sizeof(int)))) int v4i_t;
typedef __attribute__((address_space(1))) v4i_t* gv4i_p;
typedef __attribute__((address_space(3))) v4i_t* lv4i_p;

__device__ __forceinline__ void async_g2l_b128(const void* g, const void* l) {
  __builtin_amdgcn_global_load_async_to_lds_b128(
      (gv4i_p)(uintptr_t)g,
      (lv4i_p)(unsigned int)(uintptr_t)l,
      0, 0);
}
#endif

// ---------------------------------------------------------------------------
// NT GEMM: C[M,N] = act(A[M,K] @ Bt[N,K]^T) (+bias). A is fp32 (converted to
// bf16 while staging to LDS) or bf16. Requires M%128==0, N%128==0, K%32==0.
// WMODE: 0 = fp32 row-major, 1 = bf16 row-major, 2 = bf16 transposed [N][M].
// ---------------------------------------------------------------------------
#define LSTR 20  // LDS row stride in dwords (40 bf16; 80B, 16B-aligned, low conflict)

template <bool AF32, bool RELU, bool BIAS, int WMODE>
__global__ __launch_bounds__(256) void gemm_nt(
    const void* __restrict__ Ap, const unsigned short* __restrict__ Bt,
    float* __restrict__ Cf, unsigned short* __restrict__ Cb,
    const float* __restrict__ bias, int M, int N, int K) {
  __shared__ unsigned int sA[2][128 * LSTR];
  __shared__ unsigned int sB[2][128 * LSTR];

  const int tid  = threadIdx.x;
  const int wave = tid >> 5, lane = tid & 31;
  const int wm = wave & 3, wn = wave >> 2;      // 4x2 wave grid: 32x64 per wave
  const int m16 = lane & 15, hh = lane >> 4;    // wave32 halves
  const long rowBlock = (long)blockIdx.y * 128;
  const long colBlock = (long)blockIdx.x * 128;

  const float*          Af = (const float*)Ap;
  const unsigned short* Ab = (const unsigned short*)Ap;

  v8f acc[2][4];
#pragma unroll
  for (int i = 0; i < 2; ++i)
#pragma unroll
    for (int jn = 0; jn < 4; ++jn)
#pragma unroll
      for (int e = 0; e < 8; ++e) acc[i][jn][e] = 0.f;

  const int nk = K >> 5;
  float4 raf[4]; uint4 rab[2]; uint4 rbb[2];

  // ---- register-staged paths (fallback, and always for fp32 A) ----
  auto gloadA_reg = [&](int kt) {
    const long kk = (long)kt * 32;
    if (AF32) {
#pragma unroll
      for (int c = 0; c < 4; ++c) {            // 128 rows x 32 f32: 4x16B/thread
        int ch = tid + (c << 8);
        int row = ch >> 3, off = (ch & 7) << 2;
        raf[c] = *(const float4*)(Af + (rowBlock + row) * (long)K + kk + off);
      }
    } else {
#pragma unroll
      for (int c = 0; c < 2; ++c) {            // 128 rows x 32 bf16: 2x16B/thread
        int ch = tid + (c << 8);
        int row = ch >> 2, off = (ch & 3) << 3;
        rab[c] = *(const uint4*)(Ab + (rowBlock + row) * (long)K + kk + off);
      }
    }
  };
  auto sstoreA_reg = [&](int buf) {
    if (AF32) {
#pragma unroll
      for (int c = 0; c < 4; ++c) {
        int ch = tid + (c << 8);
        int row = ch >> 3, dw = (ch & 7) << 1;
        *(uint2*)&sA[buf][row * LSTR + dw] =
            make_uint2(pk2bf(raf[c].x, raf[c].y), pk2bf(raf[c].z, raf[c].w));
      }
    } else {
#pragma unroll
      for (int c = 0; c < 2; ++c) {
        int ch = tid + (c << 8);
        int row = ch >> 2, dw = (ch & 3) << 2;
        *(uint4*)&sA[buf][row * LSTR + dw] = rab[c];
      }
    }
  };
  auto gloadB_reg = [&](int kt) {
    const long kk = (long)kt * 32;
#pragma unroll
    for (int c = 0; c < 2; ++c) {
      int ch = tid + (c << 8);
      int row = ch >> 2, off = (ch & 3) << 3;
      rbb[c] = *(const uint4*)(Bt + (colBlock + row) * (long)K + kk + off);
    }
  };
  auto sstoreB_reg = [&](int buf) {
#pragma unroll
    for (int c = 0; c < 2; ++c) {
      int ch = tid + (c << 8);
      int row = ch >> 2, dw = (ch & 3) << 2;
      *(uint4*)&sB[buf][row * LSTR + dw] = rbb[c];
    }
  };

#if HAVE_ASYNC
  // ---- async global->LDS paths (bf16 operands; no VGPR staging) ----
  auto gloadB_async = [&](int kt, int buf) {
    const long kk = (long)kt * 32;
#pragma unroll
    for (int c = 0; c < 2; ++c) {
      int ch = tid + (c << 8);
      int row = ch >> 2, off = (ch & 3) << 3, dw = (ch & 3) << 2;
      async_g2l_b128(Bt + (colBlock + row) * (long)K + kk + off,
                     &sB[buf][row * LSTR + dw]);
    }
  };
  auto gloadA_async = [&](int kt, int buf) {   // only valid when !AF32
    const long kk = (long)kt * 32;
#pragma unroll
    for (int c = 0; c < 2; ++c) {
      int ch = tid + (c << 8);
      int row = ch >> 2, off = (ch & 3) << 3, dw = (ch & 3) << 2;
      async_g2l_b128(Ab + (rowBlock + row) * (long)K + kk + off,
                     &sA[buf][row * LSTR + dw]);
    }
  };
#endif

  // ---- prologue: tile 0 ----
#if HAVE_ASYNC
  if (AF32) { gloadA_reg(0); sstoreA_reg(0); } else { gloadA_async(0, 0); }
  gloadB_async(0, 0);
  __builtin_amdgcn_s_wait_asynccnt(0);
#else
  gloadA_reg(0); gloadB_reg(0);
  sstoreA_reg(0); sstoreB_reg(0);
#endif
  __syncthreads();

  int cur = 0;
  for (int kt = 0; kt < nk; ++kt) {
    const bool has = (kt + 1) < nk;
    if (has) {
#if HAVE_ASYNC
      if (AF32) gloadA_reg(kt + 1); else gloadA_async(kt + 1, cur ^ 1);
      gloadB_async(kt + 1, cur ^ 1);   // async writes go to the idle buffer
#else
      gloadA_reg(kt + 1); gloadB_reg(kt + 1);
#endif
    }

    FragU fa[2], fb[4];
#pragma unroll
    for (int i = 0; i < 2; ++i) {
      const unsigned int* p = &sA[cur][(wm * 32 + i * 16 + m16) * LSTR];
#pragma unroll
      for (int j = 0; j < 8; ++j) {
        // A 16x32 bf16 layout: VGPR j holds K = (j<4 ? 2j : 16+2(j-4)) + 8*hh
        int dw = (j & 3) + ((j & 4) << 1) + (hh << 2);
        fa[i].u[j] = p[dw];
      }
    }
#pragma unroll
    for (int jn = 0; jn < 4; ++jn) {
      const unsigned int* p = &sB[cur][(wn * 64 + jn * 16 + m16) * LSTR];
#pragma unroll
      for (int j = 0; j < 8; ++j)
        fb[jn].u[j] = p[(j & 3) + ((j & 4) << 1) + (hh << 2)];  // B 32x16: VGPR j holds K = (j<4 ? 2j : 16+2(j-4)) + 8*hh
    }
#pragma unroll
    for (int i = 0; i < 2; ++i)
#pragma unroll
      for (int jn = 0; jn < 4; ++jn)
        acc[i][jn] = __builtin_amdgcn_wmma_f32_16x16x32_bf16(
            false, fa[i].v, false, fb[jn].v, (short)0, acc[i][jn], false, false);

    if (has) {
#if HAVE_ASYNC
      if (AF32) sstoreA_reg(cur ^ 1);
#else
      sstoreA_reg(cur ^ 1); sstoreB_reg(cur ^ 1);
#endif
    }
#if HAVE_ASYNC
    __builtin_amdgcn_s_wait_asynccnt(0);  // async writes visible before barrier
#endif
    __syncthreads();
    cur ^= 1;
  }

  // Epilogue. D layout: element j -> row (j + 8*hh), col = lane&15.
#pragma unroll
  for (int i = 0; i < 2; ++i) {
    const long rbase = rowBlock + wm * 32 + i * 16 + (hh << 3);
#pragma unroll
    for (int jn = 0; jn < 4; ++jn) {
      const long cc = colBlock + wn * 64 + jn * 16 + m16;
      const float b = BIAS ? bias[cc] : 0.f;
      float vv[8];
#pragma unroll
      for (int j = 0; j < 8; ++j) {
        float v = acc[i][jn][j] + b;
        if (RELU) v = fmaxf(v, 0.f);
        vv[j] = v;
      }
      if (WMODE == 0) {
        // fp32 row-major: 16 lanes cover 64B per row -> coalesced b32 stores
#pragma unroll
        for (int j = 0; j < 8; ++j) Cf[(rbase + j) * (long)N + cc] = vv[j];
      } else if (WMODE == 1) {
        // bf16 row-major
#pragma unroll
        for (int j = 0; j < 8; ++j) store_bf16(&Cb[(rbase + j) * (long)N + cc], vv[j]);
      } else {
        // bf16 transposed [N][M]: the 8 values are 8 consecutive rows at one
        // column = 16 contiguous bytes -> single b128 store per fragment.
        uint4 pkv = make_uint4(pk2bf(vv[0], vv[1]), pk2bf(vv[2], vv[3]),
                               pk2bf(vv[4], vv[5]), pk2bf(vv[6], vv[7]));
        *(uint4*)(Cb + cc * (long)M + rbase) = pkv;
      }
    }
  }
}

// --- small helper kernels ---------------------------------------------------

// out[c*R + r] = bf16(in[r*C + c])  (weight transpose+convert)
__global__ void tconv_kernel(const float* __restrict__ in, unsigned short* __restrict__ out,
                             int R, int C) {
  int idx = blockIdx.x * 256 + threadIdx.x;
  if (idx >= R * C) return;
  int r = idx / C, c = idx - r * C;
  store_bf16(&out[c * R + r], in[idx]);
}

// Stage 1 of group-evidence reduction: 64 deterministic partials over 128 rows.
__global__ void reduce1_kernel(const float* __restrict__ P,
                               float* __restrict__ pprec, float* __restrict__ pmsum) {
  const int c = threadIdx.x & 127;
  const int part = (blockIdx.x << 1) + (threadIdx.x >> 7);  // 0..63
  const long r0 = (long)part * 128;
  float sp = 0.f, sm = 0.f;
  for (int i = 0; i < 128; ++i) {
    const float* row = P + (r0 + i) * 512;
    float v = expf(row[384 + c]);          // var = exp(clz_logvar)
    if (v == 0.f) v = 1e-6f;
    float inv = 1.f / v;
    sp += inv;
    sm += row[256 + c] * inv;              // clz_mu * inv
  }
  pprec[part * 128 + c] = sp;
  pmsum[part * 128 + c] = sm;
}

// Stage 2: combine partials, compute gmu/glogvar/class_lat and fold the
// lower half of Wl plus bl into a per-column constant c0.
__global__ void finalize_kernel(const float* __restrict__ pprec, const float* __restrict__ pmsum,
                                const float* __restrict__ Wl, const float* __restrict__ bl,
                                const float* __restrict__ eps_group,
                                float* __restrict__ gmu, float* __restrict__ glv,
                                float* __restrict__ c0) {
  __shared__ float cl[128];
  const int n = threadIdx.x;  // 128 threads
  float sp = 0.f, sm = 0.f;
  for (int p = 0; p < 64; ++p) { sp += pprec[p * 128 + n]; sm += pmsum[p * 128 + n]; }
  float gv = 1.f / sp;
  float mu = gv * sm;
  float gvv = (gv == 0.f) ? 1e-6f : gv;
  float lg = logf(gvv);
  gmu[n] = mu;
  glv[n] = lg;
  cl[n] = mu + expf(0.5f * lg) * eps_group[n];  // class_lat (same for all rows)
  __syncthreads();
  float s = bl[n];
  for (int k = 0; k < 128; ++k) s += cl[k] * Wl[(128 + k) * 128 + n];
  c0[n] = s;
}

// z = eps_z*exp(logvar)+mu; scatter outputs; emit bf16 z for the final GEMMs.
__global__ void ewz_kernel(const float* __restrict__ P, const float* __restrict__ eps_z,
                           const float* __restrict__ gmu, const float* __restrict__ glv,
                           float* __restrict__ oz, float* __restrict__ omu,
                           float* __restrict__ olv, float* __restrict__ ogmu,
                           float* __restrict__ oglv, unsigned short* __restrict__ zb) {
  const long idx = (long)blockIdx.x * 256 + threadIdx.x;
  const long r = idx >> 7;
  const int c = (int)(idx & 127);
  const float* row = P + r * 512;
  float mu = row[c], lv = row[128 + c];
  float z = eps_z[idx] * expf(lv) + mu;
  oz[idx] = z;
  omu[idx] = mu;
  olv[idx] = lv;
  ogmu[idx] = gmu[c];
  oglv[idx] = glv[c];
  store_bf16(&zb[idx], z);
}

// ---------------------------------------------------------------------------

extern "C" void kernel_launch(void* const* d_in, const int* in_sizes, int n_in,
                              void* d_out, int out_size, void* d_ws, size_t ws_size,
                              hipStream_t stream) {
  (void)in_sizes; (void)n_in; (void)out_size; (void)ws_size;
  const int N = 8192, F_IN = 512, H1 = 256, H2 = 128;

  const float* x   = (const float*)d_in[0];
  const float* adj = (const float*)d_in[1];
  const float* W1  = (const float*)d_in[2];
  const float* W2  = (const float*)d_in[3];
  const float* W3  = (const float*)d_in[4];
  const float* W4  = (const float*)d_in[5];
  const float* W5  = (const float*)d_in[6];
  const float* Wl  = (const float*)d_in[7];
  const float* bl  = (const float*)d_in[8];
  const float* eps_z = (const float*)d_in[9];
  const float* eps_g = (const float*)d_in[10];

  // d_out sections (fp32), reference return order.
  float* out     = (float*)d_out;
  float* o_recon = out;                                // [8192][8192]
  float* o_z     = out + (long)N * N;                  // [8192][128]
  float* o_mu    = o_z   + (long)N * H2;
  float* o_lv    = o_mu  + (long)N * H2;
  float* o_gmu   = o_lv  + (long)N * H2;
  float* o_glv   = o_gmu + (long)N * H2;

  // workspace carve (~33 MB)
  char* wp = (char*)d_ws;
  auto carve = [&](size_t bytes) -> void* {
    void* p = (void*)wp;
    wp += (bytes + 255) & ~(size_t)255;
    return p;
  };
  unsigned short* T1t   = (unsigned short*)carve((size_t)H1 * N * 2);      // [256][8192]
  unsigned short* h1t   = (unsigned short*)carve((size_t)H1 * N * 2);      // [256][8192]
  unsigned short* Amatb = (unsigned short*)carve((size_t)N * H1 * 2);      // [8192][256]
  float*          P     = (float*)carve((size_t)N * 512 * 4);              // [8192][512]
  unsigned short* zb    = (unsigned short*)carve((size_t)N * H2 * 2);
  unsigned short* Ub    = (unsigned short*)carve((size_t)N * H2 * 2);
  unsigned short* W1t   = (unsigned short*)carve((size_t)H1 * F_IN * 2);   // [256][512]
  unsigned short* WcatT = (unsigned short*)carve((size_t)512 * H1 * 2);    // [512][256]
  unsigned short* WlT   = (unsigned short*)carve((size_t)H2 * H2 * 2);     // [128][128]
  float* pprec = (float*)carve(64 * 128 * 4);
  float* pmsum = (float*)carve(64 * 128 * 4);
  float* gmu   = (float*)carve(128 * 4);
  float* glv   = (float*)carve(128 * 4);
  float* c0    = (float*)carve(128 * 4);

  // Weight transposes (Bt = [N][K] layout for the NT GEMM).
  tconv_kernel<<<(F_IN * H1 + 255) / 256, 256, 0, stream>>>(W1, W1t, F_IN, H1);
  tconv_kernel<<<(H1 * H2 + 255) / 256, 256, 0, stream>>>(W2, WcatT + 0 * H2 * H1, H1, H2);
  tconv_kernel<<<(H1 * H2 + 255) / 256, 256, 0, stream>>>(W3, WcatT + 1 * H2 * H1, H1, H2);
  tconv_kernel<<<(H1 * H2 + 255) / 256, 256, 0, stream>>>(W4, WcatT + 2 * H2 * H1, H1, H2);
  tconv_kernel<<<(H1 * H2 + 255) / 256, 256, 0, stream>>>(W5, WcatT + 3 * H2 * H1, H1, H2);
  tconv_kernel<<<(H2 * H2 + 255) / 256, 256, 0, stream>>>(Wl, WlT, H2, H2);  // top 128 rows

  // T1^T = (x @ W1)^T   : A=x fp32, Bt=W1t
  gemm_nt<true, false, false, 2><<<dim3(H1 / 128, N / 128), 256, 0, stream>>>(
      x, W1t, nullptr, T1t, nullptr, N, H1, F_IN);
  // h1^T = relu(adj @ T1)^T
  gemm_nt<true, true, false, 2><<<dim3(H1 / 128, N / 128), 256, 0, stream>>>(
      adj, T1t, nullptr, h1t, nullptr, N, H1, N);
  // Amat = adj @ h1 (bf16 row-major)
  gemm_nt<true, false, false, 1><<<dim3(H1 / 128, N / 128), 256, 0, stream>>>(
      adj, h1t, nullptr, Amatb, nullptr, N, H1, N);
  // P = Amat @ [W2|W3|W4|W5] (fp32)
  gemm_nt<false, false, false, 0><<<dim3(512 / 128, N / 128), 256, 0, stream>>>(
      Amatb, WcatT, P, nullptr, nullptr, N, 512, H1);

  // Group evidence (deterministic two-stage) + elementwise.
  reduce1_kernel<<<32, 256, 0, stream>>>(P, pprec, pmsum);
  finalize_kernel<<<1, 128, 0, stream>>>(pprec, pmsum, Wl, bl, eps_g, gmu, glv, c0);
  ewz_kernel<<<(N * H2) / 256, 256, 0, stream>>>(P, eps_z, gmu, glv,
                                                 o_z, o_mu, o_lv, o_gmu, o_glv, zb);

  // U = z @ Wl_top + c0 (bf16 row-major; c0 folds class_lat branch + bias)
  gemm_nt<false, false, true, 1><<<dim3(H2 / 128, N / 128), 256, 0, stream>>>(
      zb, WlT, nullptr, Ub, c0, N, H2, H2);
  // recon = U @ U^T (fp32 to d_out)
  gemm_nt<false, false, false, 0><<<dim3(N / 128, N / 128), 256, 0, stream>>>(
      Ub, Ub, o_recon, nullptr, nullptr, N, N, H2);
}
